// CausalSelfAttention_68109591380507
// MI455X (gfx1250) — matrix operand order repaired and checked by reading, not compiled, not run on hardware
//
#include <hip/hip_runtime.h>
#include <hip/hip_bf16.h>

typedef __attribute__((ext_vector_type(16))) __bf16 v16bf;
typedef __attribute__((ext_vector_type(8)))  __bf16 v8bf;
typedef __attribute__((ext_vector_type(8)))  float  v8f;
typedef __attribute__((ext_vector_type(4)))  unsigned int u32x4;
typedef __attribute__((ext_vector_type(8)))  int i32x8;
typedef __attribute__((ext_vector_type(4)))  int i32x4;

#define BATCH  4
#define SEQ    2048
#define DMODEL 1024
#define NHEAD  16
#define HDIM   64

#define WMMA_BF16(a, b, c) \
  __builtin_amdgcn_wmma_f32_16x16x32_bf16(false, (a), false, (b), (short)0, (c), false, false)

static __device__ __forceinline__ v16bf cat8(v8bf lo, v8bf hi) {
  return __builtin_shufflevector(lo, hi, 0,1,2,3,4,5,6,7,8,9,10,11,12,13,14,15);
}

// ---------------------------------------------------------------------------
// fp32 -> bf16 conversion (vectorized x4)
// ---------------------------------------------------------------------------
__global__ __launch_bounds__(256) void cvt_bf16_kernel(const float* __restrict__ src,
                                                       __bf16* __restrict__ dst, int n) {
  int i = (blockIdx.x * blockDim.x + threadIdx.x) * 4;
  if (i >= n) return;
  float4 f = *(const float4*)(src + i);
  dst[i + 0] = (__bf16)f.x;
  dst[i + 1] = (__bf16)f.y;
  dst[i + 2] = (__bf16)f.z;
  dst[i + 3] = (__bf16)f.w;
}

// ---------------------------------------------------------------------------
// bf16 GEMM: C(MxN) = A(MxK) @ B(KxN), fp32 accumulate, WMMA 16x16x32 bf16.
// Block tile 128x64, 8 waves x (32x32), K-step 32, DOUBLE-BUFFERED LDS:
// next tile's global loads issue before compute, drain to alternate buffer
// after the WMMAs, one barrier per K-step. global_prefetch 2 steps ahead.
// MODE 0: scatter into Q/K/V bf16 (B,H,S,hd) + bias. MODE 1: fp32 out + bias.
// ---------------------------------------------------------------------------
template <int MODE>
__global__ __launch_bounds__(256) void gemm_kernel(
    const __bf16* __restrict__ A, const __bf16* __restrict__ Bm,
    const float* __restrict__ bias,
    __bf16* __restrict__ qout, __bf16* __restrict__ kout, __bf16* __restrict__ vout,
    float* __restrict__ out, int M, int N, int K) {
  __shared__ __align__(32) __bf16 As[2][128 * 32];   // A tiles (M x Kstep)
  __shared__ __align__(32) __bf16 BsT[2][64 * 32];   // B tiles transposed (N x Kstep)

  const int tid  = threadIdx.x;
  const int wave = tid >> 5, lane = tid & 31;
  const int half = lane >> 4, l16 = lane & 15;
  const int wm = wave >> 1, wn = wave & 1;
  const int m0 = blockIdx.y * 128, n0 = blockIdx.x * 64;

  // per-thread staging coordinates
  const int ar0 = (tid * 8) >> 5,        ac0 = (tid * 8) & 31;        // A chunk 0
  const int ar1 = ((tid + 256) * 8) >> 5, ac1 = ((tid + 256) * 8) & 31; // A chunk 1
  const int br  = (tid * 8) >> 6,        bc  = (tid * 8) & 63;        // B chunk

  uint4 areg0, areg1, breg;
  auto gload = [&](int kk) {
    areg0 = *(const uint4*)(A + (size_t)(m0 + ar0) * K + kk + ac0);
    areg1 = *(const uint4*)(A + (size_t)(m0 + ar1) * K + kk + ac1);
    breg  = *(const uint4*)(Bm + (size_t)(kk + br) * N + n0 + bc);
  };
  auto lstore = [&](int buf) {
    *(uint4*)(&As[buf][ar0 * 32 + ac0]) = areg0;
    *(uint4*)(&As[buf][ar1 * 32 + ac1]) = areg1;
    const __bf16* e8 = (const __bf16*)&breg;
#pragma unroll
    for (int j = 0; j < 8; j++) BsT[buf][(bc + j) * 32 + br] = e8[j];
  };

  v8f acc[2][2] = {};

  gload(0);
  lstore(0);
  __syncthreads();

  for (int k0 = 0; k0 < K; k0 += 32) {
    const int cur = (k0 >> 5) & 1;
    const bool has_next = (k0 + 32) < K;
    if (has_next) gload(k0 + 32);                 // in flight during compute
    if (k0 + 64 < K) {                            // L2 prefetch 2 steps ahead
      __builtin_prefetch(A + (size_t)(m0 + ar0) * K + k0 + 64 + ac0, 0, 0);
      __builtin_prefetch(Bm + (size_t)(k0 + 64 + br) * N + n0 + bc, 0, 0);
    }

    // fragments: A and B layout {k 0-7,16-23}/{8-15,24-31}
    v16bf af[2], bfr[2];
#pragma unroll
    for (int sm = 0; sm < 2; sm++) {
      const __bf16* p = &As[cur][(wm * 32 + sm * 16 + l16) * 32 + half * 8];
      af[sm] = cat8(*(const v8bf*)p, *(const v8bf*)(p + 16));
    }
#pragma unroll
    for (int sn = 0; sn < 2; sn++) {
      bfr[sn] = cat8(*(const v8bf*)(&BsT[cur][(wn * 32 + sn * 16 + l16) * 32 + half * 8]), *(const v8bf*)(&BsT[cur][(wn * 32 + sn * 16 + l16) * 32 + half * 8 + 16]));
    }
#pragma unroll
    for (int sm = 0; sm < 2; sm++)
#pragma unroll
      for (int sn = 0; sn < 2; sn++)
        acc[sm][sn] = WMMA_BF16(af[sm], bfr[sn], acc[sm][sn]);

    if (has_next) lstore(1 - cur);                // drain global loads into alt buffer
    __syncthreads();
  }

  // epilogue (C layout: row = e + 8*half, col = l16)
#pragma unroll
  for (int sm = 0; sm < 2; sm++)
#pragma unroll
    for (int sn = 0; sn < 2; sn++)
#pragma unroll
      for (int e = 0; e < 8; e++) {
        int row = m0 + wm * 32 + sm * 16 + e + half * 8;
        int col = n0 + wn * 32 + sn * 16 + l16;
        float val = acc[sm][sn][e] + bias[col];
        if (MODE == 0) {
          int b = row >> 11, s = row & 2047;
          int part = col >> 10, dmod = col & 1023, h = dmod >> 6, dd = dmod & 63;
          __bf16* dst = (part == 0) ? qout : (part == 1) ? kout : vout;
          dst[(((size_t)b * NHEAD + h) * SEQ + s) * HDIM + dd] = (__bf16)val;
        } else {
          out[(size_t)row * DMODEL + col] = val;
        }
      }
}

// ---------------------------------------------------------------------------
// TDM: issue tensor_load_to_lds of a 32x64 bf16 tile (rows x HDIM, row-major)
// from global (tile start = gaddr) into LDS at lds_off. D# built per ISA 8.3/8.4.
// ---------------------------------------------------------------------------
static __device__ __forceinline__ void tdm_load_tile_32x64(unsigned lds_off,
                                                           unsigned long long gaddr) {
  u32x4 g0 = {
      1u,                                        // count=1, user mode
      lds_off,                                   // lds_addr (bytes)
      (unsigned)(gaddr & 0xffffffffu),           // global_addr[31:0]
      (unsigned)((gaddr >> 32) & 0x1ffffffu) | (2u << 30)  // addr[56:32] | type=2
  };
  i32x8 g1 = {
      (int)(1u << 16),          // workgroup_mask=0, data_size=1 (2 bytes)
      (int)((unsigned)HDIM << 16),   // tensor_dim0[15:0]=64 (bits 63:48)
      (int)((unsigned)SEQ << 16),    // tensor_dim1[15:0]=2048 (bits 95:80)
      (int)((unsigned)HDIM << 16),   // tile_dim0=64 (bits 127:112)
      32,                            // tile_dim1=32, tile_dim2=0
      HDIM,                          // tensor_dim0_stride=64 (low 32 of 48)
      0, 0                           // stride0 hi / stride1
  };
  i32x4 gz = {0, 0, 0, 0};
#if defined(__clang_major__) && (__clang_major__ >= 23)
  i32x8 gz8 = {0, 0, 0, 0, 0, 0, 0, 0};
  __builtin_amdgcn_tensor_load_to_lds(g0, g1, gz, gz, gz8, 0);
#else
  __builtin_amdgcn_tensor_load_to_lds(g0, g1, gz, gz, 0);
#endif
}

// ---------------------------------------------------------------------------
// Flash attention: grid (S/64, B*H), 128 threads (4 waves x 16 query rows).
// K tile staged by the Tensor Data Mover (TENSORcnt); V tile staged manually
// transposed (TDM cannot transpose). Online softmax fp32; WMMA bf16.
// ---------------------------------------------------------------------------
__global__ __launch_bounds__(128) void flash_kernel(
    const __bf16* __restrict__ qb, const __bf16* __restrict__ kb,
    const __bf16* __restrict__ vb, __bf16* __restrict__ attn) {
  __shared__ __align__(32) __bf16 Ks[32 * 64];    // keys x dims (B-layout for K^T)
  __shared__ __align__(32) __bf16 Vt[64 * 32];    // dims x keys (B-layout for V)
  __shared__ __align__(32) __bf16 Ps[4][16 * 32]; // per-wave P restage

  const int tid  = threadIdx.x;
  const int wave = tid >> 5, lane = tid & 31;
  const int half = lane >> 4, l16 = lane & 15;
  const int bh = blockIdx.y, b = bh >> 4, h = bh & 15;
  const int qbase = blockIdx.x * 64;
  const int qrow0 = qbase + wave * 16;

  const __bf16* qp = qb + (size_t)(b * NHEAD + h) * SEQ * HDIM;
  const __bf16* kp = kb + (size_t)(b * NHEAD + h) * SEQ * HDIM;
  const __bf16* vp = vb + (size_t)(b * NHEAD + h) * SEQ * HDIM;
  const unsigned ks_lds = (unsigned)(size_t)(&Ks[0]);

  // Q fragments (16 rows x 64 dims = two 16x32 A-fragments), registers
  v16bf aq[2];
#pragma unroll
  for (int c = 0; c < 2; c++) {
    const __bf16* base = qp + (qrow0 + l16) * HDIM + c * 32 + half * 8;
    aq[c] = cat8(*(const v8bf*)base, *(const v8bf*)(base + 16));
  }

  v8f o[4] = {};
  v8f mrow, lrow;
#pragma unroll
  for (int e = 0; e < 8; e++) { mrow[e] = -1e30f; lrow[e] = 0.0f; }

  const int nblk = (qbase + 64) >> 5;
  for (int kblk = 0; kblk < nblk; kblk++) {
    // --- K tile via Tensor Data Mover (wave 0 issues; all waves barrier)
    if (wave == 0) {
      tdm_load_tile_32x64(ks_lds,
                          (unsigned long long)(size_t)(kp + (size_t)kblk * 32 * HDIM));
    }
    // --- V tile: manual transpose staging (dims-major for B-fragments)
#pragma unroll
    for (int it = 0; it < 2; it++) {
      int idx = (tid + it * 128) * 8;
      int r = idx >> 6, c = idx & 63;
      uint4 d = *(const uint4*)(vp + (kblk * 32 + r) * HDIM + c);
      const __bf16* e8 = (const __bf16*)&d;
#pragma unroll
      for (int j = 0; j < 8; j++) Vt[(c + j) * 32 + r] = e8[j];
    }
    if (wave == 0) __builtin_amdgcn_s_wait_tensorcnt(0);
    __syncthreads();

    // --- scores: 16 queries x 32 keys (two C tiles), K-dim 64 (two WMMA each)
    v8f sc[2] = {};
#pragma unroll
    for (int nt = 0; nt < 2; nt++)
#pragma unroll
      for (int c = 0; c < 2; c++) {
        v16bf bf = cat8(*(const v8bf*)(Ks + (nt * 16 + l16) * 64 + c * 32 + half * 8), *(const v8bf*)(Ks + (nt * 16 + l16) * 64 + c * 32 + half * 8 + 16));
        sc[nt] = WMMA_BF16(aq[c], bf, sc[nt]);
      }

    // --- scale + causal mask (row = e + 8*half, col = l16 + 16*nt)
#pragma unroll
    for (int nt = 0; nt < 2; nt++)
#pragma unroll
      for (int e = 0; e < 8; e++) {
        int qg = qrow0 + e + half * 8;
        int kg = kblk * 32 + nt * 16 + l16;
        float s = sc[nt][e] * 0.125f;  // 1/sqrt(64)
        sc[nt][e] = (kg > qg) ? -1e30f : s;
      }

    // --- row max across the 16 lanes of each half-wave
    v8f rmax;
#pragma unroll
    for (int e = 0; e < 8; e++) rmax[e] = fmaxf(sc[0][e], sc[1][e]);
    for (int off = 1; off <= 8; off *= 2)
#pragma unroll
      for (int e = 0; e < 8; e++)
        rmax[e] = fmaxf(rmax[e], __shfl_xor(rmax[e], off, 32));

    v8f mnew, alpha, rsum;
#pragma unroll
    for (int e = 0; e < 8; e++) {
      mnew[e]  = fmaxf(mrow[e], rmax[e]);
      alpha[e] = __expf(mrow[e] - mnew[e]);
    }
#pragma unroll
    for (int nt = 0; nt < 2; nt++)
#pragma unroll
      for (int e = 0; e < 8; e++) sc[nt][e] = __expf(sc[nt][e] - mnew[e]);
#pragma unroll
    for (int e = 0; e < 8; e++) rsum[e] = sc[0][e] + sc[1][e];
    for (int off = 1; off <= 8; off *= 2)
#pragma unroll
      for (int e = 0; e < 8; e++) rsum[e] += __shfl_xor(rsum[e], off, 32);
#pragma unroll
    for (int e = 0; e < 8; e++) {
      lrow[e] = lrow[e] * alpha[e] + rsum[e];
      mrow[e] = mnew[e];
    }
#pragma unroll
    for (int c = 0; c < 4; c++)
#pragma unroll
      for (int e = 0; e < 8; e++) o[c][e] *= alpha[e];

    // --- restage P (C-layout) into per-wave LDS, reload as A-layout
    __bf16* P = &Ps[wave][0];
#pragma unroll
    for (int nt = 0; nt < 2; nt++)
#pragma unroll
      for (int e = 0; e < 8; e++)
        P[(e + half * 8) * 32 + nt * 16 + l16] = (__bf16)sc[nt][e];
    asm volatile("s_wait_dscnt 0" ::: "memory");  // wave-internal LDS RAW

    v16bf ap = cat8(*(const v8bf*)(P + l16 * 32 + half * 8),
                    *(const v8bf*)(P + l16 * 32 + half * 8 + 16));
#pragma unroll
    for (int c = 0; c < 4; c++) {
      v16bf bv = cat8(*(const v8bf*)(Vt + (c * 16 + l16) * 32 + half * 8), *(const v8bf*)(Vt + (c * 16 + l16) * 32 + half * 8 + 16));
      o[c] = WMMA_BF16(ap, bv, o[c]);
    }
    __syncthreads();
  }

  // --- normalize and store attn (B,S,D) bf16
  v8f inv;
#pragma unroll
  for (int e = 0; e < 8; e++) inv[e] = 1.0f / lrow[e];
#pragma unroll
  for (int c = 0; c < 4; c++)
#pragma unroll
    for (int e = 0; e < 8; e++) {
      int s = qrow0 + e + half * 8;
      attn[((size_t)b * SEQ + s) * DMODEL + h * HDIM + c * 16 + l16] =
          (__bf16)(o[c][e] * inv[e]);
    }
}

// ---------------------------------------------------------------------------
extern "C" void kernel_launch(void* const* d_in, const int* in_sizes, int n_in,
                              void* d_out, int out_size, void* d_ws, size_t ws_size,
                              hipStream_t stream) {
  const float* x      = (const float*)d_in[0];
  // d_in[1] = mask: causality applied analytically, tensor unused
  const float* w_attn = (const float*)d_in[2];
  const float* b_attn = (const float*)d_in[3];
  const float* w_proj = (const float*)d_in[4];
  const float* b_proj = (const float*)d_in[5];
  float* out = (float*)d_out;

  char* ws = (char*)d_ws;
  __bf16* xb    = (__bf16*)(ws);                        // 16 MB
  __bf16* wqkvb = (__bf16*)(ws + (size_t)16 * 1048576); //  6 MB
  __bf16* wprjb = (__bf16*)(ws + (size_t)22 * 1048576); //  2 MB
  __bf16* qb    = (__bf16*)(ws + (size_t)24 * 1048576); // 16 MB
  __bf16* kb    = (__bf16*)(ws + (size_t)40 * 1048576); // 16 MB
  __bf16* vb    = (__bf16*)(ws + (size_t)56 * 1048576); // 16 MB
  __bf16* attnb = (__bf16*)(ws + (size_t)72 * 1048576); // 16 MB  (total 88 MB)

  const int NX  = BATCH * SEQ * DMODEL;     // 8388608
  const int NWA = DMODEL * 3 * DMODEL;      // 3145728
  const int NWP = DMODEL * DMODEL;          // 1048576

  cvt_bf16_kernel<<<NX / 1024, 256, 0, stream>>>(x, xb, NX);
  cvt_bf16_kernel<<<NWA / 1024, 256, 0, stream>>>(w_attn, wqkvb, NWA);
  cvt_bf16_kernel<<<NWP / 1024, 256, 0, stream>>>(w_proj, wprjb, NWP);

  dim3 g0(3 * DMODEL / 64, (BATCH * SEQ) / 128);  // (48, 64)
  gemm_kernel<0><<<g0, 256, 0, stream>>>(xb, wqkvb, b_attn, qb, kb, vb, nullptr,
                                         BATCH * SEQ, 3 * DMODEL, DMODEL);

  dim3 gf(SEQ / 64, BATCH * NHEAD);               // (32, 64)
  flash_kernel<<<gf, 128, 0, stream>>>(qb, kb, vb, attnb);

  dim3 g1(DMODEL / 64, (BATCH * SEQ) / 128);      // (16, 64)
  gemm_kernel<1><<<g1, 256, 0, stream>>>(attnb, wprjb, b_proj, nullptr, nullptr, nullptr,
                                         out, BATCH * SEQ, DMODEL, DMODEL);
}
